// BaseGAT_60318520705757
// MI455X (gfx1250) — matrix operand order repaired and checked by reading, not compiled, not run on hardware
//
#include <hip/hip_runtime.h>
#include <hip/hip_bf16.h>
#include <stdint.h>

typedef __attribute__((ext_vector_type(16))) _Float16 v16h;
typedef __attribute__((ext_vector_type(8)))  float    v8f;

#define NN 100000
#define NE 1600000

// ---------- float <-> order-preserving uint (for atomic segment-max) ----------
__device__ __forceinline__ unsigned fkey(float f) {
    unsigned u = __float_as_uint(f);
    return (u & 0x80000000u) ? ~u : (u | 0x80000000u);
}
__device__ __forceinline__ float fdec(unsigned k) {
    unsigned u = (k & 0x80000000u) ? (k & 0x7fffffffu) : ~k;
    return __uint_as_float(u);
}

// ---------- pack [W_l | W_r] -> f16 in WMMA-fragment-ready (swizzled) order ----------
// Layout: wh[(((k0/32)*nT + tile)*32 + lane)*16 + j] = B[k][col] with
//   col = tile*16 + (lane&15),  k = k0 + (lane<16 ? 0 : 8) + (j&7) + (j<8 ? 0 : 16)
// so each lane's 16 B-halves for one k-step are 32 contiguous bytes (2x b128).
__global__ void pack_w_kernel(const float* __restrict__ Wl, const float* __restrict__ Wr,
                              const float* __restrict__ bl, const float* __restrict__ br,
                              _Float16* __restrict__ wh, float* __restrict__ bias,
                              int K, int N, int Nh) {
    int i = blockIdx.x * blockDim.x + threadIdx.x;
    if (i >= K * N) return;
    int nT   = N >> 4;
    int j    = i & 15;
    int lane = (i >> 4) & 31;
    int t    = i >> 9;              // kblock*nT + tile
    int tile = t % nT;
    int kb   = t / nT;
    int col  = tile * 16 + (lane & 15);
    int k    = kb * 32 + ((lane < 16) ? 0 : 8) + (j & 7) + ((j & 8) ? 16 : 0);
    float w  = (col < Nh) ? Wl[k * Nh + col] : Wr[k * Nh + (col - Nh)];
    wh[i] = (_Float16)w;
    if (i < N) bias[i] = (i < Nh) ? bl[i] : br[i - Nh];
}

// ---------- f32 -> f16 ----------
__global__ void conv_f16_kernel(const float* __restrict__ in, _Float16* __restrict__ out, int n) {
    int i = blockIdx.x * blockDim.x + threadIdx.x;
    if (i < n) out[i] = (_Float16)in[i];
}

// ---------- WMMA GEMM: C[M,N] = A[M,K](f16) * Bpacked(f16) + bias[N]; wave = 16x16 tile ----------
template<int K>
__global__ __launch_bounds__(256)
void gemm_wmma_kernel(const _Float16* __restrict__ A, const _Float16* __restrict__ Bp,
                      const float* __restrict__ bias, float* __restrict__ C,
                      int M, int N) {
    int gw   = (blockIdx.x * blockDim.x + threadIdx.x) >> 5;
    int lane = threadIdx.x & 31;
    int nT = N >> 4, mT = M >> 4;
    if (gw >= mT * nT) return;                 // whole wave exits -> EXEC all-1s for WMMA
    int tm = gw / nT, tn = gw - tm * nT;

    int rowA  = lane & 15;                     // A: lanes 0-15 hold K 0..7 & 16..23; lanes 16-31 K 8..15 & 24..31
    int kAoff = (lane < 16) ? 0 : 8;

    v8f acc = {};
    const _Float16* Arow = A + (size_t)(tm * 16 + rowA) * K + kAoff;
    #pragma unroll
    for (int k0 = 0; k0 < K; k0 += 32) {
        v16h a, b;
        #pragma unroll
        for (int j = 0; j < 8; ++j) a[j] = Arow[k0 + j];           // contiguous -> b128
        #pragma unroll
        for (int j = 0; j < 8; ++j) a[8 + j] = Arow[k0 + 16 + j];  // contiguous -> b128
        const _Float16* bp = Bp + ((((size_t)(k0 >> 5)) * nT + tn) * 32 + lane) * 16;
        #pragma unroll
        for (int j = 0; j < 16; ++j) b[j] = bp[j];                 // contiguous -> 2x b128
        acc = __builtin_amdgcn_wmma_f32_16x16x32_f16(false, a, false, b,
                                                     (short)0, acc, false, false);
    }
    int rbase = (lane < 16) ? 0 : 8, ccol = lane & 15;
    int cg = tn * 16 + ccol;
    #pragma unroll
    for (int i = 0; i < 8; ++i)
        C[(size_t)(tm * 16 + rbase + i) * N + cg] = acc[i] + bias[cg];
}

// ---------- edge logits + atomic segment max; one wave per edge ----------
// CPL channels/lane, LPH lanes/head; HC = 32*CPL; C=64
template<int CPL, int LPH>
__global__ __launch_bounds__(256)
void edge_logit_kernel(const float* __restrict__ xlxr, int ldx, int xroff,
                       const long long* __restrict__ src, const long long* __restrict__ dst,
                       const float* __restrict__ eattr, const float* __restrict__ We,
                       const float* __restrict__ att,
                       float* __restrict__ logits, unsigned* __restrict__ nmax,
                       int E, int H) {
    int e    = (blockIdx.x * blockDim.x + threadIdx.x) >> 5;
    int lane = threadIdx.x & 31;
    if (e >= E) return;
    int s = (int)src[e], d = (int)dst[e];
    float ea = eattr[e];
    const float* xl = xlxr + (size_t)s * ldx;
    const float* xr = xlxr + (size_t)d * ldx + xroff;
    int c0 = lane * CPL;
    float part = 0.f;
    #pragma unroll
    for (int j = 0; j < CPL; ++j) {
        int c = c0 + j;
        float t = xl[c] + xr[c] + ea * We[c];
        t = (t > 0.f) ? t : 0.2f * t;            // leaky_relu(0.2)
        part += t * att[c];
    }
    #pragma unroll
    for (int m = 1; m < LPH; m <<= 1) part += __shfl_xor(part, m, 32);
    if ((lane & (LPH - 1)) == 0) {
        int head = lane / LPH;
        logits[(size_t)e * H + head] = part;
        atomicMax(&nmax[(size_t)d * H + head], fkey(part));
    }
}

// ---------- exp(logit - max[dst]) in place, atomic segment sum ----------
__global__ void edge_exp_kernel(float* __restrict__ lg, const long long* __restrict__ dst,
                                const unsigned* __restrict__ nmax, float* __restrict__ denom,
                                int E, int H) {
    int i = blockIdx.x * blockDim.x + threadIdx.x;
    if (i >= E * H) return;
    int e = i / H, h = i - e * H;
    int d = (int)dst[e];
    float ex = __expf(lg[i] - fdec(nmax[(size_t)d * H + h]));
    lg[i] = ex;
    atomicAdd(&denom[(size_t)d * H + h], ex);
}

// ---------- weighted scatter-add: agg[dst] += xl[src] * alpha; one wave per edge ----------
template<int CPL, int LPH>
__global__ __launch_bounds__(256)
void edge_agg_kernel(const float* __restrict__ xlxr, int ldx,
                     const float* __restrict__ ex, const float* __restrict__ denom,
                     const long long* __restrict__ src, const long long* __restrict__ dst,
                     float* __restrict__ agg, int E, int H) {
    int e    = (blockIdx.x * blockDim.x + threadIdx.x) >> 5;
    int lane = threadIdx.x & 31;
    if (e >= E) return;
    int s = (int)src[e], d = (int)dst[e];
    int head = lane / LPH;
    float alpha = ex[(size_t)e * H + head] / denom[(size_t)d * H + head];
    const float* xl = xlxr + (size_t)s * ldx;
    float* ag = agg + (size_t)d * (32 * CPL);
    int c0 = lane * CPL;
    #pragma unroll
    for (int j = 0; j < CPL; ++j)
        atomicAdd(&ag[c0 + j], xl[c0 + j] * alpha);
}

// ---------- h = relu(agg + bias1) -> f16 ----------
__global__ void node_relu_f16_kernel(const float* __restrict__ agg, const float* __restrict__ bias,
                                     _Float16* __restrict__ out, int n, int F) {
    int i = blockIdx.x * blockDim.x + threadIdx.x;
    if (i >= n) return;
    float v = agg[i] + bias[i % F];
    out[i] = (_Float16)(v > 0.f ? v : 0.f);
}

// ---------- out = agg2 + bias2 ----------
__global__ void final_bias_kernel(const float* __restrict__ agg, const float* __restrict__ bias,
                                  float* __restrict__ out, int n, int F) {
    int i = blockIdx.x * blockDim.x + threadIdx.x;
    if (i >= n) return;
    out[i] = agg[i] + bias[i % F];
}

static inline size_t alignup(size_t v) { return (v + 255) & ~(size_t)255; }

extern "C" void kernel_launch(void* const* d_in, const int* in_sizes, int n_in,
                              void* d_out, int out_size, void* d_ws, size_t ws_size,
                              hipStream_t stream) {
    const float*      x     = (const float*)d_in[0];
    const long long*  eidx  = (const long long*)d_in[1];
    const long long*  esrc  = eidx;
    const long long*  edst  = eidx + NE;
    const float*      eattr = (const float*)d_in[2];
    const float *W1l = (const float*)d_in[3],  *b1l = (const float*)d_in[4];
    const float *W1r = (const float*)d_in[5],  *b1r = (const float*)d_in[6];
    const float *W1e = (const float*)d_in[7],  *att1 = (const float*)d_in[8];
    const float *bias1 = (const float*)d_in[9];
    const float *W2l = (const float*)d_in[10], *b2l = (const float*)d_in[11];
    const float *W2r = (const float*)d_in[12], *b2r = (const float*)d_in[13];
    const float *W2e = (const float*)d_in[14], *att2 = (const float*)d_in[15];
    const float *bias2 = (const float*)d_in[16];

    // workspace layout
    char* base = (char*)d_ws;
    size_t o = 0;
    _Float16* xh   = (_Float16*)(base + o); o = alignup(o + (size_t)NN * 128 * 2);   // 25.6 MB
    _Float16* wh   = (_Float16*)(base + o); o = alignup(o + 128 * 256 * 2);          // 64 KB
    float*    bcmb = (float*)   (base + o); o = alignup(o + 256 * 4);
    float*    xlxr = (float*)   (base + o); o = alignup(o + (size_t)NN * 256 * 4);   // 102.4 MB
    float*    lgts = (float*)   (base + o); o = alignup(o + (size_t)NE * 2 * 4);     // 12.8 MB
    unsigned* nmax = (unsigned*)(base + o); o = alignup(o + (size_t)NN * 2 * 4);
    float*    den  = (float*)   (base + o); o = alignup(o + (size_t)NN * 2 * 4);
    float*    agg  = (float*)   (base + o); o = alignup(o + (size_t)NN * 128 * 4);   // 51.2 MB

    const int TPB = 256;
    const int edgeWaveBlocks = (NE * 32 + TPB - 1) / TPB;   // one wave per edge

    // ===================== layer 1 =====================
    pack_w_kernel<<<(128 * 256 + TPB - 1) / TPB, TPB, 0, stream>>>(W1l, W1r, b1l, b1r, wh, bcmb, 128, 256, 128);
    conv_f16_kernel<<<((size_t)NN * 128 + TPB - 1) / TPB, TPB, 0, stream>>>(x, xh, NN * 128);

    {   // xlxr[100k,256] = xh @ [W1l|W1r] + [b1l|b1r]
        int waves = (NN / 16) * (256 / 16);
        gemm_wmma_kernel<128><<<(waves * 32 + TPB - 1) / TPB, TPB, 0, stream>>>(xh, wh, bcmb, xlxr, NN, 256);
    }

    hipMemsetAsync(nmax, 0, (size_t)NN * 2 * 4, stream);   // key 0 == below -inf
    hipMemsetAsync(den,  0, (size_t)NN * 2 * 4, stream);
    hipMemsetAsync(agg,  0, (size_t)NN * 128 * 4, stream);

    edge_logit_kernel<4, 16><<<edgeWaveBlocks, TPB, 0, stream>>>(
        xlxr, 256, 128, esrc, edst, eattr, W1e, att1, lgts, nmax, NE, 2);
    edge_exp_kernel<<<((size_t)NE * 2 + TPB - 1) / TPB, TPB, 0, stream>>>(lgts, edst, nmax, den, NE, 2);
    edge_agg_kernel<4, 16><<<edgeWaveBlocks, TPB, 0, stream>>>(
        xlxr, 256, lgts, den, esrc, edst, agg, NE, 2);

    node_relu_f16_kernel<<<((size_t)NN * 128 + TPB - 1) / TPB, TPB, 0, stream>>>(agg, bias1, xh, NN * 128, 128);

    // ===================== layer 2 =====================
    pack_w_kernel<<<(128 * 128 + TPB - 1) / TPB, TPB, 0, stream>>>(W2l, W2r, b2l, b2r, wh, bcmb, 128, 128, 64);
    {   // xlxr[100k,128] = h @ [W2l|W2r] + [b2l|b2r]
        int waves = (NN / 16) * (128 / 16);
        gemm_wmma_kernel<128><<<(waves * 32 + TPB - 1) / TPB, TPB, 0, stream>>>(xh, wh, bcmb, xlxr, NN, 128);
    }

    hipMemsetAsync(nmax, 0, (size_t)NN * 4, stream);
    hipMemsetAsync(den,  0, (size_t)NN * 4, stream);
    hipMemsetAsync(agg,  0, (size_t)NN * 64 * 4, stream);

    edge_logit_kernel<2, 32><<<edgeWaveBlocks, TPB, 0, stream>>>(
        xlxr, 128, 64, esrc, edst, eattr, W2e, att2, lgts, nmax, NE, 1);
    edge_exp_kernel<<<((size_t)NE + TPB - 1) / TPB, TPB, 0, stream>>>(lgts, edst, nmax, den, NE, 1);
    edge_agg_kernel<2, 32><<<edgeWaveBlocks, TPB, 0, stream>>>(
        xlxr, 128, lgts, den, esrc, edst, agg, NE, 1);

    final_bias_kernel<<<((size_t)NN * 64 + TPB - 1) / TPB, TPB, 0, stream>>>(agg, bias2, (float*)d_out, NN * 64, 64);
}
